// Dense_test_40510131536145
// MI455X (gfx1250) — compile-verified, matrix operand order checked by reading, not run on hardware
//
#include <hip/hip_runtime.h>

// DH-SNN Dense_test forward, MI455X (gfx1250), wave32 WMMA.
// B=512, T=2000, N_IN=32, N_HID=200, N_OUT=3, BRANCH=1, VTH=1.0

typedef _Float16 v16h __attribute__((ext_vector_type(16)));
typedef _Float16 v8h  __attribute__((ext_vector_type(8)));
typedef float    v8f  __attribute__((ext_vector_type(8)));

#define B_TOT    512
#define T_STEPS  2000
#define NIN      32
#define NHID     200
#define NOUT     3
#define FAN_IN   232          // NIN + NHID
#define K_PAD    256          // K padded to 8 chunks of 32
#define NT       13           // 13 N-tiles of 16 cover 208 >= 200 hidden units
#define WG_THREADS (NT * 32)  // 416 threads = 13 wave32

__device__ __forceinline__ float sigmoidf_(float v) {
    return 1.0f / (1.0f + __expf(-v));
}

// Workgroup barrier that only drains the LDS counter (dscnt), NOT loadcnt.
// __syncthreads() would wait loadcnt==0 too, killing the x(t+1) software
// pipeline. LDS producer/consumer ordering across waves needs only dscnt.
__device__ __forceinline__ void wg_sync_lds() {
    asm volatile(
        "s_wait_dscnt 0x0\n\t"
        "s_barrier_signal -1\n\t"
        "s_barrier_wait -1"
        ::: "memory");
}

__global__ __launch_bounds__(WG_THREADS)
void dhsnn_dense_kernel(const float* __restrict__ x,      // [B][T][NIN]
                        const float* __restrict__ W1,     // [NHID][FAN_IN]
                        const float* __restrict__ b1,     // [NHID]
                        const float* __restrict__ tau_n,  // [NHID] (branch=1)
                        const float* __restrict__ tau_m1, // [NHID]
                        const float* __restrict__ W2,     // [NOUT][NHID]
                        const float* __restrict__ b2,     // [NOUT]
                        const float* __restrict__ tau_m2, // [NOUT]
                        const float* __restrict__ mask,   // [NHID][FAN_IN] (all ones)
                        float* __restrict__ out)          // [B][NOUT]
{
    __shared__ __align__(16) _Float16 Albs[16][K_PAD]; // [batch-row][k]  k = [x(32) | spk(200) | pad]
    __shared__ __align__(16) float W2lds[NOUT][NHID];

    const int tid   = threadIdx.x;
    const int wave  = tid >> 5;
    const int lane  = tid & 31;
    const int m     = lane & 15;     // A-matrix row / C-matrix column index within tile
    const int hi    = lane >> 4;     // half-wave selector for K packing
    const int h     = wave * 16 + m; // hidden unit owned by this lane (B/C column)
    const int btile = blockIdx.x * 16;

    // ---- one-time setup --------------------------------------------------
    for (int i = tid; i < 16 * K_PAD; i += WG_THREADS)
        ((_Float16*)Albs)[i] = (_Float16)0.0f;           // pad columns stay zero forever
    for (int i = tid; i < NOUT * NHID; i += WG_THREADS)
        W2lds[i / NHID][i % NHID] = W2[i];

    const bool hvalid = (h < NHID);

    // Preload B fragments (masked weights, f32 -> f16), one v16h per K-chunk.
    // 16-bit layout: lane holds column n=h; hi=0 -> K {base..base+7, base+16..base+23}
    v16h bfrag[8];
    for (int kc = 0; kc < 8; ++kc) {
        v16h bf;
        const int base = kc * 32 + hi * 8;
        for (int j = 0; j < 16; ++j) {
            const int k = base + ((j < 8) ? j : (8 + j)); // j>=8 -> base+16+(j-8)
            float wv = 0.0f;
            if (hvalid && k < FAN_IN)
                wv = W1[h * FAN_IN + k] * mask[h * FAN_IN + k];
            bf[j] = (_Float16)wv;
        }
        bfrag[kc] = bf;
    }

    // per-lane neuron parameters (h is uniform across the 8 C-fragment VGPRs)
    float beta = 0.0f, alpha1 = 0.0f;
    v8f cbias;
    {
        float bias1 = 0.0f;
        if (hvalid) {
            beta   = sigmoidf_(tau_n[h]);
            alpha1 = sigmoidf_(tau_m1[h]);
            bias1  = b1[h];
        }
        for (int r = 0; r < 8; ++r) cbias[r] = bias1;
    }

    // readout threads: tid < 48 -> (batch row rb, output rowo)
    const bool ro   = (tid < 16 * NOUT);
    const int  rb   = tid / NOUT;
    const int  rowo = tid % NOUT;
    float alpha2 = 0.0f, bias2 = 0.0f, mem2 = 0.0f, acc = 0.0f;
    if (ro) { alpha2 = sigmoidf_(tau_m2[rowo]); bias2 = b2[rowo]; }

    // neuron state in C-fragment layout: VGPR r holds batch row (hi?8+r:r), column h
    v8f d_in, mem1, spk;
    for (int r = 0; r < 8; ++r) { d_in[r] = 0.0f; mem1[r] = 0.0f; spk[r] = 0.0f; }

    // software pipeline for x: hold x(t) in registers a full step ahead
    const bool xth = (tid < 256);
    const int  xrow = (tid * 2) >> 5;
    const int  xcol = (tid * 2) & 31;
    const size_t xbase = (((size_t)(btile + xrow)) * T_STEPS) * NIN + xcol;
    float2 xreg = make_float2(0.0f, 0.0f);
    if (xth) xreg = *(const float2*)&x[xbase];            // t = 0

    __syncthreads();  // setup fence (full fence OK outside the hot loop)

    // ---- time loop (serial dependency chain) -----------------------------
    for (int t = 0; t < T_STEPS; ++t) {
        // stage x(t) from registers into A columns 0..31 (f32 -> f16)
        if (xth) {
            Albs[xrow][xcol]     = (_Float16)xreg.x;
            Albs[xrow][xcol + 1] = (_Float16)xreg.y;
        }
        // issue the HBM load for x(t+1) now; it has the whole step to land
        float2 xnext = xreg;
        if (xth && (t + 1 < T_STEPS))
            xnext = *(const float2*)&x[xbase + (size_t)(t + 1) * NIN];
        wg_sync_lds();   // dscnt-only: does NOT drain the x(t+1) load

        // cur = [x_t, spk_prev] @ (W1*mask)^T + b1 : 8 chained WMMAs over K,
        // double-buffered A-fragment LDS loads (load kc+1 ahead of WMMA kc)
        v8f c = cbias;
        v8h lo0 = *(const v8h*)&Albs[m][hi * 8];
        v8h hh0 = *(const v8h*)&Albs[m][hi * 8 + 16];
        for (int kc = 0; kc < 8; ++kc) {
            v8h lo1 = lo0, hh1 = hh0;
            if (kc < 7) {
                const int nb = (kc + 1) * 32 + hi * 8;
                lo1 = *(const v8h*)&Albs[m][nb];
                hh1 = *(const v8h*)&Albs[m][nb + 16];
            }
            v16h a;
            for (int j = 0; j < 8; ++j) { a[j] = lo0[j]; a[8 + j] = hh0[j]; }
            c = __builtin_amdgcn_wmma_f32_16x16x32_f16(
                    /*neg_a=*/false, a, /*neg_b=*/false, bfrag[kc],
                    /*c_mod=*/(short)0, c, /*reuse_a=*/false, /*reuse_b=*/false);
            lo0 = lo1; hh0 = hh1;
        }
        wg_sync_lds();   // all A reads done before spikes are overwritten

        // elementwise neuron update (registers only); soft reset, Heaviside spike
        for (int r = 0; r < 8; ++r) {
            d_in[r] = beta   * d_in[r] + (1.0f - beta)   * c[r];
            mem1[r] = alpha1 * mem1[r] + (1.0f - alpha1) * d_in[r] - 1.0f * spk[r];
            spk[r]  = (mem1[r] - 1.0f > 0.0f) ? 1.0f : 0.0f;
        }
        // publish new spikes into A columns 32..231 for the next step
        if (hvalid) {
            const int mbase = hi * 8;
            for (int r = 0; r < 8; ++r)
                Albs[mbase + r][NIN + h] = (_Float16)spk[r];
        }
        wg_sync_lds();   // spikes visible to readout and to next step's WMMA

        // leaky readout integrator: mem2 = a2*mem2 + (1-a2)*(spk @ W2^T + b2)
        if (ro) {
            float y = bias2;
            const v8h* sp = (const v8h*)&Albs[rb][NIN];   // 200 spikes = 25 x v8h
            for (int c8 = 0; c8 < 25; ++c8) {
                const v8h s = sp[c8];
                const float* w = &W2lds[rowo][c8 * 8];
                for (int j = 0; j < 8; ++j)
                    y += (float)s[j] * w[j];
            }
            mem2 = alpha2 * mem2 + (1.0f - alpha2) * y;
            if (t > 0) acc += mem2; // reference accumulates mem2_seq[1:]
        }

        xreg = xnext;
    }

    if (ro)
        out[(btile + rb) * NOUT + rowo] = acc / (float)T_STEPS;
}

extern "C" void kernel_launch(void* const* d_in, const int* in_sizes, int n_in,
                              void* d_out, int out_size, void* d_ws, size_t ws_size,
                              hipStream_t stream) {
    (void)in_sizes; (void)n_in; (void)out_size; (void)d_ws; (void)ws_size;
    const float* x      = (const float*)d_in[0];
    const float* W1     = (const float*)d_in[1];
    const float* b1     = (const float*)d_in[2];
    const float* tau_n  = (const float*)d_in[3];
    const float* tau_m1 = (const float*)d_in[4];
    const float* W2     = (const float*)d_in[5];
    const float* b2     = (const float*)d_in[6];
    const float* tau_m2 = (const float*)d_in[7];
    const float* mask   = (const float*)d_in[8];
    float* out = (float*)d_out;

    dim3 grid(B_TOT / 16);       // 32 workgroups, one per 16-row batch tile
    dim3 block(WG_THREADS);      // 13 wave32 (one per 16-wide hidden tile)
    dhsnn_dense_kernel<<<grid, block, 0, stream>>>(
        x, W1, b1, tau_n, tau_m1, W2, b2, tau_m2, mask, out);
}
